// Torch_17377437680118
// MI455X (gfx1250) — matrix operand order repaired and checked by reading, not compiled, not run on hardware
//
#include <hip/hip_runtime.h>

#define H_DIM 2048
#define I_DIM 8192
#define T_TOK 8192

#define BM 128
#define BN 64
#define BK 32
#define LDA 40   // halves: 32 + 8 pad, 80B rows keep 16B alignment
#define LDB 40
#define NY_TILES (T_TOK / BM + 1)

#if __has_builtin(__builtin_amdgcn_global_load_async_to_lds_b128)
#define USE_ASYNC 1
#else
#define USE_ASYNC 0
#endif

typedef __attribute__((ext_vector_type(16))) __bf16 v16bf;
typedef __attribute__((ext_vector_type(2)))  __bf16 v2bf;
typedef __attribute__((ext_vector_type(8)))  float  v8f;
typedef __attribute__((ext_vector_type(4)))  int    v4i;

struct Q2 { uint4 a, b; };

__device__ __forceinline__ v16bf frag_cast(uint4 a, uint4 b) {
  Q2 q{a, b};
  return __builtin_bit_cast(v16bf, q);
}

// fp32 pair -> packed bf16 (RNE). Single V_CVT_PK_BF16_F32 when available.
__device__ __forceinline__ unsigned int pack2_bf16(float x, float y) {
#if __has_builtin(__builtin_amdgcn_cvt_pk_bf16_f32)
  v2bf r = __builtin_amdgcn_cvt_pk_bf16_f32(x, y);
  return __builtin_bit_cast(unsigned int, r);
#else
  unsigned ux = __float_as_uint(x), uy = __float_as_uint(y);
  ux = (ux + 0x7FFFu + ((ux >> 16) & 1u)) >> 16;
  uy = (uy + 0x7FFFu + ((uy >> 16) & 1u)) >> 16;
  return (ux & 0xFFFFu) | (uy << 16);
#endif
}
// scalar f32 -> bf16 in low 16 bits (b16 store consumes the low half)
__device__ __forceinline__ unsigned short f2bf(float x) {
#if __has_builtin(__builtin_amdgcn_cvt_pk_bf16_f32)
  return (unsigned short)(pack2_bf16(x, x) & 0xFFFFu);
#else
  unsigned u = __float_as_uint(x);
  return (unsigned short)((u + 0x7FFFu + ((u >> 16) & 1u)) >> 16);
#endif
}

#if USE_ASYNC
// 16B global -> LDS DMA copy (gfx1250 GLOBAL_LOAD_ASYNC_TO_LDS_B128, ASYNCcnt).
// Param type per hipcc diagnostic: pointer to 4xi32 vector (global / shared AS).
// AS1 bit pattern == generic global address; AS3 bit pattern == low 32 bits of
// the generic LDS address (flat->local addrspacecast is a truncation).
__device__ __forceinline__ void async_ld16(const unsigned short* g, unsigned short* l) {
  __builtin_amdgcn_global_load_async_to_lds_b128(
      (__attribute__((address_space(1))) v4i*)(unsigned long long)g,
      (__attribute__((address_space(3))) v4i*)(unsigned)(unsigned long long)l,
      0, 0);
}
__device__ __forceinline__ void wait_async0() {
#if __has_builtin(__builtin_amdgcn_s_wait_asynccnt)
  __builtin_amdgcn_s_wait_asynccnt(0);
#else
  asm volatile("s_wait_asynccnt 0x0" ::: "memory");
#endif
}
#endif

// ---------------------------------------------------------------------------
// Kernel 1: block-wide scan of mask -> permutation (masked tokens first).
// ---------------------------------------------------------------------------
__global__ __launch_bounds__(1024) void scan_mask(const int* __restrict__ mask,
                                                  int* __restrict__ perm,
                                                  int* __restrict__ n1p) {
  __shared__ int cnt[1024];
  const int t = threadIdx.x;
  const int base = t * 8;
  int bits[8];
  int c = 0;
#pragma unroll
  for (int j = 0; j < 8; ++j) { bits[j] = (mask[base + j] != 0); c += bits[j]; }
  cnt[t] = c;
  __syncthreads();
  for (int off = 1; off < 1024; off <<= 1) {
    int v = cnt[t];
    if (t >= off) v += cnt[t - off];
    __syncthreads();
    cnt[t] = v;
    __syncthreads();
  }
  const int total = cnt[1023];
  const int excl  = cnt[t] - c;
  if (t == 0) *n1p = total;
  int p1 = excl;
  int p0 = total + base - excl;
#pragma unroll
  for (int j = 0; j < 8; ++j) {
    if (bits[j]) perm[p1++] = base + j;
    else         perm[p0++] = base + j;
  }
}

// ---------------------------------------------------------------------------
// Kernel 2: gather x rows into permuted order, fp32 -> bf16 (RNE).
// ---------------------------------------------------------------------------
__global__ __launch_bounds__(256) void gather_cast(const float* __restrict__ x,
                                                   const int* __restrict__ perm,
                                                   unsigned short* __restrict__ xb) {
  const int tp  = blockIdx.x;
  const int src = perm[tp];
  const float4* xin = (const float4*)(x + (size_t)src * H_DIM);
  uint2* o = (uint2*)(xb + (size_t)tp * H_DIM);
  for (int i = threadIdx.x; i < H_DIM / 4; i += blockDim.x) {
    float4 v = xin[i];
    o[i] = make_uint2(pack2_bf16(v.x, v.y), pack2_bf16(v.z, v.w));
  }
}

// ---------------------------------------------------------------------------
// Kernel 3: fused up-projection: proj = silu(x*W1^T) * (x*W3^T), bf16 WMMA.
// Ping-pong LDS, async A staging, register-prefetched B tiles, 1 barrier/step.
// ---------------------------------------------------------------------------
__global__ __launch_bounds__(256) void gemm_up(
    const unsigned short* __restrict__ xb,
    const float* __restrict__ w1, const float* __restrict__ w3,
    const float* __restrict__ u1, const float* __restrict__ u3,
    const int* __restrict__ n1p,
    unsigned short* __restrict__ proj) {
  __shared__ __align__(16) unsigned short sA[2][BM * LDA];
  __shared__ __align__(16) unsigned short sB1[2][BN * LDB];
  __shared__ __align__(16) unsigned short sB3[2][BN * LDB];

  const int N1  = *n1p;
  const int nt1 = (N1 + BM - 1) / BM;
  const int nt2 = (T_TOK - N1 + BM - 1) / BM;
  const int ty  = blockIdx.y;
  int base, rows;
  const float *W1, *W3;
  if (ty < nt1) {
    base = ty * BM; rows = N1 - base; if (rows > BM) rows = BM;
    W1 = w1; W3 = w3;
  } else {
    const int j = ty - nt1;
    if (j >= nt2) return;
    base = N1 + j * BM; rows = T_TOK - base; if (rows > BM) rows = BM;
    W1 = u1; W3 = u3;
  }

  const int nBase = blockIdx.x * BN;
  const int tid   = threadIdx.x;
  const int lane  = tid & 31;
  const int wave  = tid >> 5;
  const int waveM = wave & 3;
  const int waveN = wave >> 2;
  const int l16   = lane & 15;
  const int aK0   = (lane < 16) ? 0 : 16;
  const int bK0   = (lane < 16) ? 0 : 16;

  const int aRow  = tid >> 1;
  const int aSeg  = (tid & 1) * 16;               // halves
  const int aTokR = (aRow < rows) ? aRow : (rows - 1);
  const unsigned short* aG = xb + (size_t)(base + aTokR) * H_DIM + aSeg;
  unsigned short* aL[2] = { &sA[0][aRow * LDA + aSeg], &sA[1][aRow * LDA + aSeg] };

  const int bRow = tid >> 2;
  const int bCol = (tid & 3) * 8;                 // floats
  const float* b1G = W1 + (size_t)(nBase + bRow) * H_DIM + bCol;
  const float* b3G = W3 + (size_t)(nBase + bRow) * H_DIM + bCol;
  const int bOff = bRow * LDB + bCol;

  v8f acc1[2][2] = {};
  v8f acc3[2][2] = {};

  constexpr int KT = H_DIM / BK;

  // prologue: stage k-tile 0
#if USE_ASYNC
  async_ld16(aG, aL[0]);
  async_ld16(aG + 8, aL[0] + 8);
  uint4 aR0 = {}, aR1 = {};
#else
  uint4 aR0 = ((const uint4*)aG)[0];
  uint4 aR1 = ((const uint4*)aG)[1];
#endif
  float4 r1a = ((const float4*)b1G)[0], r1b = ((const float4*)b1G)[1];
  float4 r3a = ((const float4*)b3G)[0], r3b = ((const float4*)b3G)[1];

  for (int kb = 0; kb < KT; ++kb) {
    const int cur = kb & 1;
    const int nxt = cur ^ 1;

    // commit prefetched B (and A on fallback) into buffer `cur`
    *(uint4*)&sB1[cur][bOff] = make_uint4(pack2_bf16(r1a.x, r1a.y), pack2_bf16(r1a.z, r1a.w),
                                          pack2_bf16(r1b.x, r1b.y), pack2_bf16(r1b.z, r1b.w));
    *(uint4*)&sB3[cur][bOff] = make_uint4(pack2_bf16(r3a.x, r3a.y), pack2_bf16(r3a.z, r3a.w),
                                          pack2_bf16(r3b.x, r3b.y), pack2_bf16(r3b.z, r3b.w));
#if USE_ASYNC
    wait_async0();                       // drain this wave's DMA into `cur`
#else
    ((uint4*)aL[cur])[0] = aR0;
    ((uint4*)aL[cur])[1] = aR1;
#endif
    __syncthreads();

    // issue next tile's fetches (overlaps with WMMA below)
    if (kb + 1 < KT) {
      const unsigned short* aGn = aG + (kb + 1) * BK;
#if USE_ASYNC
      async_ld16(aGn, aL[nxt]);
      async_ld16(aGn + 8, aL[nxt] + 8);
#else
      aR0 = ((const uint4*)aGn)[0];
      aR1 = ((const uint4*)aGn)[1];
#endif
      const float* g1 = b1G + (kb + 1) * BK;
      r1a = ((const float4*)g1)[0]; r1b = ((const float4*)g1)[1];
      const float* g3 = b3G + (kb + 1) * BK;
      r3a = ((const float4*)g3)[0]; r3b = ((const float4*)g3)[1];
      __builtin_prefetch(b1G + (kb + 2) * BK, 0, 1);
      __builtin_prefetch(b3G + (kb + 2) * BK, 0, 1);
    }

    const unsigned short* As  = sA[cur];
    const unsigned short* B1s = sB1[cur];
    const unsigned short* B3s = sB3[cur];

    v16bf af[2];
#pragma unroll
    for (int mt = 0; mt < 2; ++mt) {
      const unsigned short* p = &As[(waveM * 32 + mt * 16 + l16) * LDA + aK0];
      af[mt] = frag_cast(*(const uint4*)p, *(const uint4*)(p + 8));
    }
#pragma unroll
    for (int nt = 0; nt < 2; ++nt) {
      const unsigned short* p1 = &B1s[(waveN * 32 + nt * 16 + l16) * LDB + bK0];
      v16bf b1f = frag_cast(*(const uint4*)p1, *(const uint4*)(p1 + 8));
      const unsigned short* p3 = &B3s[(waveN * 32 + nt * 16 + l16) * LDB + bK0];
      v16bf b3f = frag_cast(*(const uint4*)p3, *(const uint4*)(p3 + 8));
#pragma unroll
      for (int mt = 0; mt < 2; ++mt) {
        acc1[mt][nt] = __builtin_amdgcn_wmma_f32_16x16x32_bf16(
            false, af[mt], false, b1f, (short)0, acc1[mt][nt], false, false);
        acc3[mt][nt] = __builtin_amdgcn_wmma_f32_16x16x32_bf16(
            false, af[mt], false, b3f, (short)0, acc3[mt][nt], false, false);
      }
    }
  }

  // epilogue: silu(acc1) * acc3 -> bf16 proj (permuted rows); fast v_rcp_f32
#pragma unroll
  for (int mt = 0; mt < 2; ++mt) {
    const int mBase = waveM * 32 + mt * 16 + (lane >> 4) * 8;
#pragma unroll
    for (int nt = 0; nt < 2; ++nt) {
      const int n = nBase + waveN * 32 + nt * 16 + l16;
      v8f a1 = acc1[mt][nt];
      v8f a3 = acc3[mt][nt];
#pragma unroll
      for (int r = 0; r < 8; ++r) {
        const int m = mBase + r;
        if (m < rows) {
          float g = a1[r];
          float s = g * __builtin_amdgcn_rcpf(1.0f + __expf(-g));
          proj[(size_t)(base + m) * I_DIM + (size_t)n] = f2bf(s * a3[r]);
        }
      }
    }
  }
}

// ---------------------------------------------------------------------------
// Kernel 4: down-projection: out[perm[t], h] = proj[t, :] . W2[h, :]
// ---------------------------------------------------------------------------
__global__ __launch_bounds__(256) void gemm_down(
    const unsigned short* __restrict__ proj,
    const float* __restrict__ w2, const float* __restrict__ u2,
    const int* __restrict__ n1p, const int* __restrict__ perm,
    float* __restrict__ out) {
  __shared__ __align__(16) unsigned short sA[2][BM * LDA];
  __shared__ __align__(16) unsigned short sB[2][BN * LDB];

  const int N1  = *n1p;
  const int nt1 = (N1 + BM - 1) / BM;
  const int nt2 = (T_TOK - N1 + BM - 1) / BM;
  const int ty  = blockIdx.y;
  int base, rows;
  const float* W2;
  if (ty < nt1) {
    base = ty * BM; rows = N1 - base; if (rows > BM) rows = BM;
    W2 = w2;
  } else {
    const int j = ty - nt1;
    if (j >= nt2) return;
    base = N1 + j * BM; rows = T_TOK - base; if (rows > BM) rows = BM;
    W2 = u2;
  }

  const int nBase = blockIdx.x * BN;   // over H
  const int tid   = threadIdx.x;
  const int lane  = tid & 31;
  const int wave  = tid >> 5;
  const int waveM = wave & 3;
  const int waveN = wave >> 2;
  const int l16   = lane & 15;
  const int aK0   = (lane < 16) ? 0 : 16;
  const int bK0   = (lane < 16) ? 0 : 16;

  const int aRow  = tid >> 1;
  const int aSeg  = (tid & 1) * 16;
  const int aTokR = (aRow < rows) ? aRow : (rows - 1);
  const unsigned short* aG = proj + (size_t)(base + aTokR) * I_DIM + aSeg;
  unsigned short* aL[2] = { &sA[0][aRow * LDA + aSeg], &sA[1][aRow * LDA + aSeg] };

  const int bRow = tid >> 2;
  const int bCol = (tid & 3) * 8;
  const float* bG = W2 + (size_t)(nBase + bRow) * I_DIM + bCol;
  const int bOff = bRow * LDB + bCol;

  v8f acc[2][2] = {};

  constexpr int KT = I_DIM / BK;

#if USE_ASYNC
  async_ld16(aG, aL[0]);
  async_ld16(aG + 8, aL[0] + 8);
  uint4 aR0 = {}, aR1 = {};
#else
  uint4 aR0 = ((const uint4*)aG)[0];
  uint4 aR1 = ((const uint4*)aG)[1];
#endif
  float4 rba = ((const float4*)bG)[0], rbb = ((const float4*)bG)[1];

  for (int kb = 0; kb < KT; ++kb) {
    const int cur = kb & 1;
    const int nxt = cur ^ 1;

    *(uint4*)&sB[cur][bOff] = make_uint4(pack2_bf16(rba.x, rba.y), pack2_bf16(rba.z, rba.w),
                                         pack2_bf16(rbb.x, rbb.y), pack2_bf16(rbb.z, rbb.w));
#if USE_ASYNC
    wait_async0();
#else
    ((uint4*)aL[cur])[0] = aR0;
    ((uint4*)aL[cur])[1] = aR1;
#endif
    __syncthreads();

    if (kb + 1 < KT) {
      const unsigned short* aGn = aG + (kb + 1) * BK;
#if USE_ASYNC
      async_ld16(aGn, aL[nxt]);
      async_ld16(aGn + 8, aL[nxt] + 8);
#else
      aR0 = ((const uint4*)aGn)[0];
      aR1 = ((const uint4*)aGn)[1];
#endif
      const float* g2 = bG + (kb + 1) * BK;
      rba = ((const float4*)g2)[0]; rbb = ((const float4*)g2)[1];
      __builtin_prefetch(bG + (kb + 2) * BK, 0, 1);
    }

    const unsigned short* As = sA[cur];
    const unsigned short* Bs = sB[cur];

    v16bf af[2];
#pragma unroll
    for (int mt = 0; mt < 2; ++mt) {
      const unsigned short* p = &As[(waveM * 32 + mt * 16 + l16) * LDA + aK0];
      af[mt] = frag_cast(*(const uint4*)p, *(const uint4*)(p + 8));
    }
#pragma unroll
    for (int nt = 0; nt < 2; ++nt) {
      const unsigned short* pb = &Bs[(waveN * 32 + nt * 16 + l16) * LDB + bK0];
      v16bf bf = frag_cast(*(const uint4*)pb, *(const uint4*)(pb + 8));
#pragma unroll
      for (int mt = 0; mt < 2; ++mt) {
        acc[mt][nt] = __builtin_amdgcn_wmma_f32_16x16x32_bf16(
            false, af[mt], false, bf, (short)0, acc[mt][nt], false, false);
      }
    }
  }

#pragma unroll
  for (int mt = 0; mt < 2; ++mt) {
    const int mBase = waveM * 32 + mt * 16 + (lane >> 4) * 8;
    int dstRow[8];
#pragma unroll
    for (int r = 0; r < 8; ++r) {
      const int m = mBase + r;
      dstRow[r] = (m < rows) ? perm[base + m] : -1;
    }
#pragma unroll
    for (int nt = 0; nt < 2; ++nt) {
      const int n = nBase + waveN * 32 + nt * 16 + l16;
      v8f a = acc[mt][nt];
#pragma unroll
      for (int r = 0; r < 8; ++r) {
        if (dstRow[r] >= 0) out[(size_t)dstRow[r] * H_DIM + n] = a[r];
      }
    }
  }
}

// ---------------------------------------------------------------------------
extern "C" void kernel_launch(void* const* d_in, const int* in_sizes, int n_in,
                              void* d_out, int out_size, void* d_ws, size_t ws_size,
                              hipStream_t stream) {
  (void)in_sizes; (void)n_in; (void)out_size; (void)ws_size;
  const float* x  = (const float*)d_in[0];
  const int* mask = (const int*)d_in[1];
  const float* w1 = (const float*)d_in[2];
  const float* w2 = (const float*)d_in[3];
  const float* w3 = (const float*)d_in[4];
  const float* u1 = (const float*)d_in[5];
  const float* u2 = (const float*)d_in[6];
  const float* u3 = (const float*)d_in[7];
  float* out = (float*)d_out;

  char* ws = (char*)d_ws;
  int* n1p  = (int*)ws;
  int* perm = (int*)(ws + 256);
  unsigned short* xb   = (unsigned short*)(ws + 65536);
  unsigned short* proj = (unsigned short*)(ws + 65536 + (size_t)T_TOK * H_DIM * 2);

  scan_mask<<<1, 1024, 0, stream>>>(mask, perm, n1p);
  gather_cast<<<T_TOK, 256, 0, stream>>>(x, perm, xb);
  gemm_up<<<dim3(I_DIM / BN, NY_TILES), 256, 0, stream>>>(xb, w1, w3, u1, u3, n1p, proj);
  gemm_down<<<dim3(H_DIM / BN, NY_TILES), 256, 0, stream>>>(proj, w2, u2, n1p, perm, out);
}
